// SlotRung_37658273252052
// MI455X (gfx1250) — matrix operand order repaired and checked by reading, not compiled, not run on hardware
//
#include <hip/hip_runtime.h>
#include <hip/hip_bf16.h>
#include <math.h>

typedef __attribute__((ext_vector_type(16))) __bf16 v16bf;
typedef __attribute__((ext_vector_type(8)))  float  v8f;

#define NB      32768
#define SSLOT   8
#define KD      96
#define VD      192
#define CTXD    1024
#define CHD     512
#define GHD     384
#define MTD     1024
#define GATEK   1701
#define GATEKP  1728      // GATEK padded to multiple of 32
#define TAILW   704
#define T_CK    0
#define T_CV    96
#define T_MV    288
#define T_PV    480
#define T_SC    672   // best_score, matched_conf, cadence, surprise, incoming

__device__ __forceinline__ float wsum32(float v) {
#pragma unroll
  for (int o = 16; o > 0; o >>= 1) v += __shfl_xor(v, o, 32);
  return v;
}
__device__ __forceinline__ float sigm(float x) { return 1.0f / (1.0f + __expf(-x)); }
__device__ __forceinline__ float gelu(float x) { return 0.5f * x * (1.0f + erff(x * 0.70710678118654752f)); }

// A fragment loader: 16 bf16 from two aligned 8-float groups (ISA A layout)
__device__ __forceinline__ v16bf load_a(const float* __restrict__ r, int abase) {
  v16bf a;
#pragma unroll
  for (int g = 0; g < 2; ++g) {
    const float4 f0 = *reinterpret_cast<const float4*>(r + g * 16 + abase);
    const float4 f1 = *reinterpret_cast<const float4*>(r + g * 16 + abase + 4);
    a[g * 8 + 0] = (__bf16)f0.x; a[g * 8 + 1] = (__bf16)f0.y;
    a[g * 8 + 2] = (__bf16)f0.z; a[g * 8 + 3] = (__bf16)f0.w;
    a[g * 8 + 4] = (__bf16)f1.x; a[g * 8 + 5] = (__bf16)f1.y;
    a[g * 8 + 6] = (__bf16)f1.z; a[g * 8 + 7] = (__bf16)f1.w;
  }
  return a;
}

// ---------------------------------------------------------------------------
// Weight pre-pack: W[K,N] f32 row-major -> bf16 fragments in WMMA B layout:
//   Wp[((ntile*nKtot + ks)*32 + lane)*16 + h] = W[ks*32 + (lane>>4)*8 + (h&7) + (h>>3)*16][ntile*16 + (lane&15)]
// Zero-fills kw >= Kreal, so GEMM inner loops need no guards.
// ---------------------------------------------------------------------------
__global__ __launch_bounds__(256) void pack_w(
    const float* __restrict__ W, int Kreal, int N,
    int nKtot, int ntileOff, int nLocalTiles, __bf16* __restrict__ out)
{
  const int idx = blockIdx.x * 256 + threadIdx.x;
  if (idx >= nLocalTiles * nKtot * 32) return;
  const int lane = idx & 31;
  const int ks   = (idx >> 5) % nKtot;
  const int lt   = (idx >> 5) / nKtot;
  const int col  = lt * 16 + (lane & 15);
  const int kb   = ks * 32 + (lane >> 4) * 8;
  __bf16* o = out + ((size_t)((ntileOff + lt) * nKtot + ks) * 32 + lane) * 16;
#pragma unroll
  for (int h = 0; h < 16; ++h) {
    const int kw = kb + (h & 7) + ((h >> 3) << 4);
    const float v = (kw < Kreal) ? W[(size_t)kw * N + col] : 0.0f;
    o[h] = (__bf16)v;
  }
}

// ---------------------------------------------------------------------------
// Activation pre-pack: A[M,K] f32 row-major -> bf16 fragments in WMMA A layout
// (16-row tiles). One thread converts one 32-byte lane fragment.
// ---------------------------------------------------------------------------
__global__ __launch_bounds__(256) void pack_a(
    const float* __restrict__ A, int lda, int nK, int totalFrag,
    __bf16* __restrict__ out)
{
  const int idx = blockIdx.x * 256 + threadIdx.x;
  if (idx >= totalFrag) return;
  const int lane = idx & 31;
  const int ks   = (idx >> 5) % nK;
  const int mt   = (idx >> 5) / nK;
  const float* r = A + (size_t)(mt * 16 + (lane & 15)) * lda + ks * 32;
  const v16bf a = load_a(r, (lane >> 4) * 8);
  *reinterpret_cast<v16bf*>(out + ((size_t)(mt * nK + ks) * 32 + lane) * 16) = a;
}

__global__ __launch_bounds__(256) void copyf_kernel(
    const float* __restrict__ src, float* __restrict__ dst, int n)
{
  const int i = blockIdx.x * 256 + threadIdx.x;
  if (i < n) dst[i] = src[i];
}

// ---------------------------------------------------------------------------
// Mixed-precision GEMM: C[M,N] = act(A[M,K] @ W[K,N] + bias[N])
// K range [0, ksplit) comes from A1 (pre-packed bf16 fragments if PA1, else
// f32 rows converted on load); [ksplit, Kpad) comes from f32 A2. W is the
// pre-packed bf16 fragment buffer. One wave computes a 32x32 output block:
// 2x2 sub-tiles, 4 v_wmma_f32_16x16x32_bf16 per 32-wide K step, no guards.
// ---------------------------------------------------------------------------
template <bool PA1>
__global__ __launch_bounds__(256) void gemm_bf16(
    const void* __restrict__ A1v, int lda1, int nKa1, int ksplit,
    const float* __restrict__ A2, int lda2,
    const __bf16* __restrict__ Wp, const float* __restrict__ bias,
    float* __restrict__ C, int ldc, int M, int N, int Kpad, int act)
{
  const int lane = threadIdx.x & 31;
  const int wave = threadIdx.x >> 5;
  const int t32n = N >> 5;
  const int tile = blockIdx.x * 8 + wave;
  if (tile >= (M >> 5) * t32n) return;
  const int m0    = (tile / t32n) << 5;
  const int n0    = (tile % t32n) << 5;
  const int l15   = lane & 15;
  const int half  = lane >> 4;
  const int abase = half * 8;
  const int nK    = Kpad >> 5;

  const __bf16* bp0 = Wp + ((size_t)(n0 >> 4) * nK * 32 + lane) * 16;
  const __bf16* bp1 = bp0 + (size_t)nK * 512;

  v8f c00 = {}, c01 = {}, c10 = {}, c11 = {};

  if (PA1) {
    const __bf16* A1p = (const __bf16*)A1v;
    const __bf16* ap0 = A1p + ((size_t)(m0 >> 4) * nKa1 * 32 + lane) * 16;
    const __bf16* ap1 = ap0 + (size_t)nKa1 * 512;
#pragma unroll 2
    for (int k0 = 0; k0 < ksplit; k0 += 32) {
      const size_t ks = (size_t)(k0 >> 5) * 512;
      const v16bf a0 = *reinterpret_cast<const v16bf*>(ap0 + ks);
      const v16bf a1 = *reinterpret_cast<const v16bf*>(ap1 + ks);
      const v16bf b0 = *reinterpret_cast<const v16bf*>(bp0 + ks);
      const v16bf b1 = *reinterpret_cast<const v16bf*>(bp1 + ks);
      c00 = __builtin_amdgcn_wmma_f32_16x16x32_bf16(false, a0, false, b0, (short)0, c00, false, false);
      c01 = __builtin_amdgcn_wmma_f32_16x16x32_bf16(false, a0, false, b1, (short)0, c01, false, false);
      c10 = __builtin_amdgcn_wmma_f32_16x16x32_bf16(false, a1, false, b0, (short)0, c10, false, false);
      c11 = __builtin_amdgcn_wmma_f32_16x16x32_bf16(false, a1, false, b1, (short)0, c11, false, false);
    }
  } else {
    const float* A1 = (const float*)A1v;
    const float* a1r0 = A1 + (size_t)(m0 + l15) * lda1;
    const float* a1r1 = A1 + (size_t)(m0 + 16 + l15) * lda1;
#pragma unroll 2
    for (int k0 = 0; k0 < ksplit; k0 += 32) {
      const v16bf a0 = load_a(a1r0 + k0, abase);
      const v16bf a1 = load_a(a1r1 + k0, abase);
      const size_t ks = (size_t)(k0 >> 5) * 512;
      const v16bf b0 = *reinterpret_cast<const v16bf*>(bp0 + ks);
      const v16bf b1 = *reinterpret_cast<const v16bf*>(bp1 + ks);
      c00 = __builtin_amdgcn_wmma_f32_16x16x32_bf16(false, a0, false, b0, (short)0, c00, false, false);
      c01 = __builtin_amdgcn_wmma_f32_16x16x32_bf16(false, a0, false, b1, (short)0, c01, false, false);
      c10 = __builtin_amdgcn_wmma_f32_16x16x32_bf16(false, a1, false, b0, (short)0, c10, false, false);
      c11 = __builtin_amdgcn_wmma_f32_16x16x32_bf16(false, a1, false, b1, (short)0, c11, false, false);
    }
  }
  // second A source (f32 rows, convert on load)
  {
    const float* a2r0 = A2 + (size_t)(m0 + l15) * lda2;
    const float* a2r1 = A2 + (size_t)(m0 + 16 + l15) * lda2;
#pragma unroll 2
    for (int k0 = ksplit; k0 < Kpad; k0 += 32) {
      const v16bf a0 = load_a(a2r0 + (k0 - ksplit), abase);
      const v16bf a1 = load_a(a2r1 + (k0 - ksplit), abase);
      const size_t ks = (size_t)(k0 >> 5) * 512;
      const v16bf b0 = *reinterpret_cast<const v16bf*>(bp0 + ks);
      const v16bf b1 = *reinterpret_cast<const v16bf*>(bp1 + ks);
      c00 = __builtin_amdgcn_wmma_f32_16x16x32_bf16(false, a0, false, b0, (short)0, c00, false, false);
      c01 = __builtin_amdgcn_wmma_f32_16x16x32_bf16(false, a0, false, b1, (short)0, c01, false, false);
      c10 = __builtin_amdgcn_wmma_f32_16x16x32_bf16(false, a1, false, b0, (short)0, c10, false, false);
      c11 = __builtin_amdgcn_wmma_f32_16x16x32_bf16(false, a1, false, b1, (short)0, c11, false, false);
    }
  }

  const int   cb  = n0 + l15;
  const float b0v = bias[cb];
  const float b1v = bias[cb + 16];
#pragma unroll
  for (int r = 0; r < 8; ++r) {
    const int row0 = m0 + half * 8 + r;
    const int row1 = row0 + 16;
    float v00 = c00[r] + b0v, v01 = c01[r] + b1v;
    float v10 = c10[r] + b0v, v11 = c11[r] + b1v;
    if (act == 1) { v00 = gelu(v00); v01 = gelu(v01); v10 = gelu(v10); v11 = gelu(v11); }
    C[(size_t)row0 * ldc + cb]      = v00;
    C[(size_t)row0 * ldc + cb + 16] = v01;
    C[(size_t)row1 * ldc + cb]      = v10;
    C[(size_t)row1 * ldc + cb + 16] = v11;
  }
}

// ---------------------------------------------------------------------------
// Match kernel: normalize cand_key (in place in tail), cosine-match against
// live slots, gather matched value/conf/age, write gf-tail scalars + zero pad.
// One wave per batch row.
// ---------------------------------------------------------------------------
__global__ __launch_bounds__(256) void match_kernel(
    const float* __restrict__ keys, const float* __restrict__ values,
    const float* __restrict__ conf, const float* __restrict__ age,
    const float* __restrict__ alive, const float* __restrict__ inb,
    float* __restrict__ tail, int* __restrict__ bidx)
{
  const int wave = threadIdx.x >> 5, lane = threadIdx.x & 31;
  const int b = blockIdx.x * 8 + wave;
  if (b >= NB) return;
  float* t = tail + (size_t)b * TAILW;

  float c0 = t[T_CK + lane], c1 = t[T_CK + lane + 32], c2 = t[T_CK + lane + 64];
  const float ss  = wsum32(c0 * c0 + c1 * c1 + c2 * c2);
  const float inv = 1.0f / fmaxf(sqrtf(ss), 1e-6f);
  c0 *= inv; c1 *= inv; c2 *= inv;
  t[T_CK + lane] = c0; t[T_CK + lane + 32] = c1; t[T_CK + lane + 64] = c2;
  if (lane < TAILW - 677) t[677 + lane] = 0.0f;   // zero K-pad of gf tail

  float best = -1e30f, anylive = 0.0f; int bi = 0;
#pragma unroll
  for (int s = 0; s < SSLOT; ++s) {
    const float* kr = keys + ((size_t)b * SSLOT + s) * KD;
    const float av  = alive[(size_t)b * SSLOT + s];
    const float adj = (1.0f - av) * 1e-4f;
    const float x0 = kr[lane] + adj, x1 = kr[lane + 32] + adj, x2 = kr[lane + 64] + adj;
    const float n2 = wsum32(x0 * x0 + x1 * x1 + x2 * x2);
    const float iv = 1.0f / fmaxf(sqrtf(n2), 1e-6f);
    const float dt = wsum32(x0 * c0 + x1 * c1 + x2 * c2) * iv;
    const float sc = (av > 0.0f) ? dt : -10000.0f;
    if (sc > best) { best = sc; bi = s; }
    anylive = fmaxf(anylive, (av > 0.0f) ? 1.0f : 0.0f);
  }
  if (anylive == 0.0f) { best = 0.0f; bi = 0; }

  const float* vr = values + ((size_t)b * SSLOT + bi) * VD;
#pragma unroll
  for (int j = 0; j < 6; ++j) t[T_MV + lane + 32 * j] = vr[lane + 32 * j];
  if (lane == 0) {
    t[T_SC + 0] = best;
    t[T_SC + 1] = conf[(size_t)b * SSLOT + bi];
    const float ma = age[(size_t)b * SSLOT + bi];
    t[T_SC + 2] = sigm((ma - 2.0f) * 0.5f);     // cadence prior
    t[T_SC + 4] = inb[b];                        // incoming bubble
    bidx[b] = bi;
  }
}

// pred_err + surprise (one wave per row)
__global__ __launch_bounds__(256) void prederr_kernel(float* __restrict__ tail)
{
  const int wave = threadIdx.x >> 5, lane = threadIdx.x & 31;
  const int b = blockIdx.x * 8 + wave;
  if (b >= NB) return;
  float* t = tail + (size_t)b * TAILW;
  float ss = 0.0f;
#pragma unroll
  for (int j = 0; j < 6; ++j) {
    const float d = t[T_CV + lane + 32 * j] - t[T_PV + lane + 32 * j];
    ss += d * d;
  }
  ss = wsum32(ss);
  if (lane == 0) {
    const float perr = ss * (1.0f / 192.0f);
    t[T_SC + 3] = 1.0f - t[T_SC + 0] + 0.25f * perr + t[T_SC + 4];
  }
}

// gate layer-2 dot product + sigmoid (one wave per row); h has stride hstride
__global__ __launch_bounds__(256) void gatedot_kernel(
    const float* __restrict__ h, int hstride, int hoff,
    const float* __restrict__ w2, const float* __restrict__ b2,
    float* __restrict__ out)
{
  const int wave = threadIdx.x >> 5, lane = threadIdx.x & 31;
  const int b = blockIdx.x * 8 + wave;
  if (b >= NB) return;
  const float* hr = h + (size_t)b * hstride + hoff;
  float s = 0.0f;
#pragma unroll
  for (int j = 0; j < 12; ++j) { const int i = lane + 32 * j; s += hr[i] * w2[i]; }
  s = wsum32(s);
  if (lane == 0) out[b] = sigm(s + b2[0]);
}

// ---------------------------------------------------------------------------
// State update: masks, spawn selection, soft scatter write, per-slot key
// renormalize, summary reduction (LDS ds_add_f32). One 256-thread block/row,
// wave s handles slot s.
// ---------------------------------------------------------------------------
__global__ __launch_bounds__(256) void update_kernel(
    const float* __restrict__ keys, const float* __restrict__ values,
    const float* __restrict__ conf, const float* __restrict__ age,
    const float* __restrict__ alive, const float* __restrict__ evid,
    const float* __restrict__ tail, const int* __restrict__ bidx,
    const float* __restrict__ rprob, const float* __restrict__ sprob,
    const float* __restrict__ pprob,
    float* __restrict__ o_rm, float* __restrict__ o_sm, float* __restrict__ o_pm,
    float* __restrict__ o_bub, float* __restrict__ o_ne,
    float* __restrict__ o_nk, float* __restrict__ o_nv, float* __restrict__ o_nc,
    float* __restrict__ o_na, float* __restrict__ o_nal,
    float* __restrict__ o_summary)
{
  __shared__ float acc[VD];
  __shared__ float wsh;
  __shared__ float wr[SSLOT];
  const int b = blockIdx.x;
  const int t = threadIdx.x;
  const int lane = t & 31, s = t >> 5;
  const float* tl = tail + (size_t)b * TAILW;

  if (t >= 192) {              // threads 192..255: zero the reduction buffers
    const int z = t - 192;
    acc[z] = 0.0f; acc[z + 64] = 0.0f; acc[z + 128] = 0.0f;
    if (t == 192) wsh = 0.0f;
  }
  if (t == 0) {                // scalar gate / spawn logic
    float hm = 0.0f;
#pragma unroll
    for (int k = 0; k < SSLOT; ++k) hm = fmaxf(hm, alive[(size_t)b * SSLOT + k]);
    const float rr  = sigm((rprob[b] - 0.55f) * 4.0f) * hm;
    const float sr  = sigm((sprob[b] - 0.60f) * 4.0f);
    const float den = 1.0f + rr + sr;
    const float rm  = rr / den, sm = sr / den;
    const float pm  = sigm((pprob[b] - 0.50f) * 4.0f);
    const int bi = bidx[b];
    int cand = 0; float bestin = -1.0f;
#pragma unroll
    for (int k = 0; k < SSLOT; ++k) {
      const float ina = (alive[(size_t)b * SSLOT + k] < 0.5f) ? 1.0f : 0.0f;
      if (ina > bestin) { bestin = ina; cand = k; }
    }
    int repl = 0; float bu = 1e30f;
#pragma unroll
    for (int k = 0; k < SSLOT; ++k) {
      float u = conf[(size_t)b * SSLOT + k] - 0.01f * age[(size_t)b * SSLOT + k];
      if (k == bi) u = 10000.0f;
      if (u < bu) { bu = u; repl = k; }
    }
    const int spi = (bestin > 0.5f) ? cand : repl;
    float mw = 0.0f;
#pragma unroll
    for (int k = 0; k < SSLOT; ++k) {
      float w = rm * ((k == bi) ? 1.0f : 0.0f) + sm * ((k == spi) ? 1.0f : 0.0f);
      w = fminf(fmaxf(w, 0.0f), 1.0f);
      wr[k] = w; mw += w;
    }
    mw *= (1.0f / (float)SSLOT);
    const float ne = 0.9f * evid[b] + mw;
    o_rm[b] = rm; o_sm[b] = sm; o_pm[b] = pm;
    o_ne[b] = ne; o_bub[b] = fmaxf(ne - 0.2f, 0.0f);
  }
  __syncthreads();

  {
    const float  w    = wr[s];
    const float  omw  = 1.0f - w;
    const size_t base = (size_t)b * SSLOT + s;
    const float* kr = keys + base * KD;
    float k0 = omw * kr[lane]      + w * tl[T_CK + lane];
    float k1 = omw * kr[lane + 32] + w * tl[T_CK + lane + 32];
    float k2 = omw * kr[lane + 64] + w * tl[T_CK + lane + 64];
    const float n2 = wsum32(k0 * k0 + k1 * k1 + k2 * k2);
    const float iv = 1.0f / fmaxf(sqrtf(n2), 1e-6f);
    o_nk[base * KD + lane]      = k0 * iv;
    o_nk[base * KD + lane + 32] = k1 * iv;
    o_nk[base * KD + lane + 64] = k2 * iv;

    const float nc  = omw * conf[base] + w;
    const float na  = fminf(fmaxf(alive[base] * 0.995f + w, 0.0f), 1.0f);
    const float wgt = nc * na;
    const float* vr = values + base * VD;
#pragma unroll
    for (int j = 0; j < 6; ++j) {
      const int i = lane + 32 * j;
      const float nv = omw * vr[i] + w * tl[T_CV + i];
      o_nv[base * VD + i] = nv;
      atomicAdd(&acc[i], nv * wgt);
    }
    if (lane == 0) {
      o_nc[base]  = nc;
      o_na[base]  = (age[base] + 1.0f) * omw;
      o_nal[base] = na;
      atomicAdd(&wsh, wgt);
    }
  }
  __syncthreads();

  if (t < VD) o_summary[(size_t)b * VD + t] = acc[t] / (wsh + 1e-6f);
}

// layernorm(pre_ln) * promote_mask -> promoted_summary (one block per row)
__global__ __launch_bounds__(256) void ln_kernel(
    const float* __restrict__ x, const float* __restrict__ g,
    const float* __restrict__ be, const float* __restrict__ pm,
    float* __restrict__ out)
{
  __shared__ float s1[256], s2[256];
  const int b = blockIdx.x, t = threadIdx.x;
  const float4 v = reinterpret_cast<const float4*>(x + (size_t)b * MTD)[t];
  s1[t] = v.x + v.y + v.z + v.w;
  s2[t] = v.x * v.x + v.y * v.y + v.z * v.z + v.w * v.w;
  __syncthreads();
  for (int o = 128; o > 0; o >>= 1) {
    if (t < o) { s1[t] += s1[t + o]; s2[t] += s2[t + o]; }
    __syncthreads();
  }
  const float mean = s1[0] * (1.0f / (float)MTD);
  const float var  = s2[0] * (1.0f / (float)MTD) - mean * mean;
  const float rstd = rsqrtf(var + 1e-5f);
  const float p    = pm[b];
  const float xv[4] = { v.x, v.y, v.z, v.w };
#pragma unroll
  for (int j = 0; j < 4; ++j) {
    const int i = t * 4 + j;
    out[(size_t)b * MTD + i] = p * ((xv[j] - mean) * rstd * g[i] + be[i]);
  }
}

// ---------------------------------------------------------------------------
extern "C" void kernel_launch(void* const* d_in, const int* in_sizes, int n_in,
                              void* d_out, int out_size, void* d_ws, size_t ws_size,
                              hipStream_t stream) {
  (void)in_sizes; (void)n_in; (void)out_size; (void)ws_size;
  const float* keys  = (const float*)d_in[0];
  const float* vals  = (const float*)d_in[1];
  const float* conf  = (const float*)d_in[2];
  const float* age   = (const float*)d_in[3];
  const float* alive = (const float*)d_in[4];
  const float* evid  = (const float*)d_in[5];
  const float* ctx   = (const float*)d_in[6];
  const float* inb   = (const float*)d_in[7];
  // params: pytree (sorted-key) order: ck, cv, pg, rg, sg, sp, vp; each MLP b1,b2,w1,w2
  const float* ck_b1 = (const float*)d_in[8];
  const float* ck_b2 = (const float*)d_in[9];
  const float* ck_w1 = (const float*)d_in[10];
  const float* ck_w2 = (const float*)d_in[11];
  const float* cv_b1 = (const float*)d_in[12];
  const float* cv_b2 = (const float*)d_in[13];
  const float* cv_w1 = (const float*)d_in[14];
  const float* cv_w2 = (const float*)d_in[15];
  const float* pg_b1 = (const float*)d_in[16];
  const float* pg_b2 = (const float*)d_in[17];
  const float* pg_w1 = (const float*)d_in[18];
  const float* pg_w2 = (const float*)d_in[19];
  const float* rg_b1 = (const float*)d_in[20];
  const float* rg_b2 = (const float*)d_in[21];
  const float* rg_w1 = (const float*)d_in[22];
  const float* rg_w2 = (const float*)d_in[23];
  const float* sg_b1 = (const float*)d_in[24];
  const float* sg_b2 = (const float*)d_in[25];
  const float* sg_w1 = (const float*)d_in[26];
  const float* sg_w2 = (const float*)d_in[27];
  const float* sp_b  = (const float*)d_in[28];
  const float* sp_be = (const float*)d_in[29];
  const float* sp_g  = (const float*)d_in[30];
  const float* sp_w  = (const float*)d_in[31];
  const float* vp_b1 = (const float*)d_in[32];
  const float* vp_b2 = (const float*)d_in[33];
  const float* vp_w1 = (const float*)d_in[34];
  const float* vp_w2 = (const float*)d_in[35];

  // ---- workspace layout ----
  float* ws    = (float*)d_ws;
  float* h     = ws;                                   // [NB, 1152] (reused at 1024/512/1152)
  float* tail  = h    + (size_t)NB * 1152;             // [NB, 704] gf tail (zero-padded)
  float* pre   = tail + (size_t)NB * TAILW;            // [NB, 1024] pre-layernorm
  float* rprob = pre  + (size_t)NB * MTD;              // [NB]
  float* sprob = rprob + NB;
  float* pprob = sprob + NB;
  int*   bidx  = (int*)(pprob + NB);                   // [NB]
  float* bias_ckcv = (float*)(bidx + NB);              // [1024] = ck_b1 | cv_b1
  float* bias_gate = bias_ckcv + 1024;                 // [1152] = rg_b1 | sg_b1 | pg_b1
  size_t poff = (size_t)((bias_gate + 1152) - ws);
  poff = (poff + 7) & ~(size_t)7;                      // 32B-align packed region
  __bf16* P_ckcv1 = (__bf16*)(ws + poff);              // 1024x1024 B fragments
  __bf16* P_ck2   = P_ckcv1 + (size_t)1024 * 1024;     //  96x512
  __bf16* P_cv2   = P_ck2   + (size_t)96 * 512;        // 192x512
  __bf16* P_vp1   = P_cv2   + (size_t)192 * 512;       // 512x1216
  __bf16* P_vp2   = P_vp1   + (size_t)512 * 1216;      // 192x512
  __bf16* P_gate  = P_vp2   + (size_t)192 * 512;       // 1152x1728
  __bf16* P_sp    = P_gate  + (size_t)1152 * 1728;     // 1024x192
  __bf16* P_ctx   = P_sp    + (size_t)1024 * 192;      // [NB,1024] packed-A context

  // ---- output layout (concatenated in reference return order) ----
  float* out    = (float*)d_out;
  float* o_prom = out;                                 // [NB,1024]
  float* o_sum  = o_prom + (size_t)NB * MTD;           // [NB,192]
  float* o_bub  = o_sum  + (size_t)NB * VD;
  float* o_rm   = o_bub  + NB;
  float* o_sm   = o_rm   + NB;
  float* o_pm   = o_sm   + NB;
  float* o_nk   = o_pm   + NB;                         // [NB,8,96]
  float* o_nv   = o_nk   + (size_t)NB * SSLOT * KD;    // [NB,8,192]
  float* o_nc   = o_nv   + (size_t)NB * SSLOT * VD;
  float* o_na   = o_nc   + (size_t)NB * SSLOT;
  float* o_nal  = o_na   + (size_t)NB * SSLOT;
  float* o_ne   = o_nal  + (size_t)NB * SSLOT;

  const dim3 blk(256);
  const int rowsGrid = NB / 8;
  const int nKctx = CTXD / 32;                         // 32
  const int ctxFrag = (NB / 16) * nKctx * 32;
#define PACK_GRID(nt, nk) dim3((((nt) * (nk) * 32) + 255) / 256)
#define GEMM_GRID(N) dim3((((NB / 32) * ((N) / 32)) + 7) / 8)

  // ---- pre-pack weights (B layout) and context (A layout), concat biases ----
  pack_a<<<dim3((ctxFrag + 255) / 256), blk, 0, stream>>>(ctx, CTXD, nKctx, ctxFrag, P_ctx);
  pack_w<<<PACK_GRID(32, 32), blk, 0, stream>>>(ck_w1, CTXD, CHD, 32, 0,  32, P_ckcv1);
  pack_w<<<PACK_GRID(32, 32), blk, 0, stream>>>(cv_w1, CTXD, CHD, 32, 32, 32, P_ckcv1);
  pack_w<<<PACK_GRID(6, 16),  blk, 0, stream>>>(ck_w2, CHD, KD, 16, 0, 6,  P_ck2);
  pack_w<<<PACK_GRID(12, 16), blk, 0, stream>>>(cv_w2, CHD, VD, 16, 0, 12, P_cv2);
  pack_w<<<PACK_GRID(32, 38), blk, 0, stream>>>(vp_w1, CTXD + VD, CHD, 38, 0, 32, P_vp1);
  pack_w<<<PACK_GRID(12, 16), blk, 0, stream>>>(vp_w2, CHD, VD, 16, 0, 12, P_vp2);
  pack_w<<<PACK_GRID(24, 54), blk, 0, stream>>>(rg_w1, GATEK, GHD, 54, 0,  24, P_gate);
  pack_w<<<PACK_GRID(24, 54), blk, 0, stream>>>(sg_w1, GATEK, GHD, 54, 24, 24, P_gate);
  pack_w<<<PACK_GRID(24, 54), blk, 0, stream>>>(pg_w1, GATEK, GHD, 54, 48, 24, P_gate);
  pack_w<<<PACK_GRID(64, 6),  blk, 0, stream>>>(sp_w, VD, MTD, 6, 0, 64, P_sp);
  copyf_kernel<<<dim3(2), blk, 0, stream>>>(ck_b1, bias_ckcv, CHD);
  copyf_kernel<<<dim3(2), blk, 0, stream>>>(cv_b1, bias_ckcv + CHD, CHD);
  copyf_kernel<<<dim3(2), blk, 0, stream>>>(rg_b1, bias_gate, GHD);
  copyf_kernel<<<dim3(2), blk, 0, stream>>>(sg_b1, bias_gate + GHD, GHD);
  copyf_kernel<<<dim3(2), blk, 0, stream>>>(pg_b1, bias_gate + 2 * GHD, GHD);

  // ---- fused ck/cv layer 1: packed context -> h[NB,1024] = [h_ck | h_cv] ----
  gemm_bf16<true><<<GEMM_GRID(1024), blk, 0, stream>>>(P_ctx, 0, nKctx, CTXD, ctx, CTXD,
      P_ckcv1, bias_ckcv, h, 1024, NB, 1024, CTXD, 1);
  // ck layer 2 -> cand_key(raw) in tail[0..96); cv layer 2 -> cand_val in tail[96..288)
  gemm_bf16<false><<<GEMM_GRID(KD), blk, 0, stream>>>(h, 1024, 0, CHD, h, 1024,
      P_ck2, ck_b2, tail + T_CK, TAILW, NB, KD, CHD, 0);
  gemm_bf16<false><<<GEMM_GRID(VD), blk, 0, stream>>>(h + CHD, 1024, 0, CHD, h + CHD, 1024,
      P_cv2, cv_b2, tail + T_CV, TAILW, NB, VD, CHD, 0);
  // match + gather (normalizes cand_key in place, zero-pads tail)
  match_kernel<<<rowsGrid, blk, 0, stream>>>(keys, vals, conf, age, alive, inb, tail, bidx);
  // vp MLP: [packed context | matched_value] -> h[NB,512] -> pred_val in tail[480..672)
  gemm_bf16<true><<<GEMM_GRID(CHD), blk, 0, stream>>>(P_ctx, 0, nKctx, CTXD, tail + T_MV, TAILW,
      P_vp1, vp_b1, h, CHD, NB, CHD, CTXD + VD, 1);
  gemm_bf16<false><<<GEMM_GRID(VD), blk, 0, stream>>>(h, CHD, 0, CHD, h, CHD,
      P_vp2, vp_b2, tail + T_PV, TAILW, NB, VD, CHD, 0);
  // pred_err -> surprise scalar in tail
  prederr_kernel<<<rowsGrid, blk, 0, stream>>>(tail);
  // fused gate MLPs layer 1: gf = [packed context | tail], K=1701 (padded 1728), N=1152
  gemm_bf16<true><<<GEMM_GRID(1152), blk, 0, stream>>>(P_ctx, 0, nKctx, CTXD, tail, TAILW,
      P_gate, bias_gate, h, 1152, NB, 1152, GATEKP, 1);
  gatedot_kernel<<<rowsGrid, blk, 0, stream>>>(h, 1152, 0,       rg_w2, rg_b2, rprob);
  gatedot_kernel<<<rowsGrid, blk, 0, stream>>>(h, 1152, GHD,     sg_w2, sg_b2, sprob);
  gatedot_kernel<<<rowsGrid, blk, 0, stream>>>(h, 1152, 2 * GHD, pg_w2, pg_b2, pprob);
  // state update + summary
  update_kernel<<<NB, blk, 0, stream>>>(keys, vals, conf, age, alive, evid,
      tail, bidx, rprob, sprob, pprob,
      o_rm, o_sm, o_pm, o_bub, o_ne, o_nk, o_nv, o_nc, o_na, o_nal, o_sum);
  // summary @ sp.w + sp.b -> pre_ln, then layernorm * promote_mask
  gemm_bf16<false><<<GEMM_GRID(MTD), blk, 0, stream>>>(o_sum, VD, 0, VD, o_sum, VD,
      P_sp, sp_b, pre, MTD, NB, MTD, VD, 0);
  ln_kernel<<<NB, blk, 0, stream>>>(pre, sp_g, sp_be, o_pm, o_prom);
#undef PACK_GRID
#undef GEMM_GRID
}
